// AttentionLayer_75471165325659
// MI455X (gfx1250) — matrix operand order repaired and checked by reading, not compiled, not run on hardware
//
#include <hip/hip_runtime.h>

typedef unsigned short u16;
typedef unsigned int   u32;
typedef __attribute__((ext_vector_type(16))) __bf16 v16bf;
typedef __attribute__((ext_vector_type(8)))  float  v8f;
typedef __attribute__((ext_vector_type(4)))  int    v4i;

// ---------------- async-to-LDS support (guarded; sync fallback) ----------------
#if defined(__has_builtin)
#if __has_builtin(__builtin_amdgcn_global_load_async_to_lds_b128) && \
    __has_builtin(__builtin_amdgcn_s_wait_asynccnt)
#define HAVE_ASYNC_LDS 1
#endif
#endif
#ifndef HAVE_ASYNC_LDS
#define HAVE_ASYNC_LDS 0
#endif

#if HAVE_ASYNC_LDS
typedef __attribute__((address_space(1))) v4i gv4i;  // global 16B vector
typedef __attribute__((address_space(3))) v4i lv4i;  // LDS 16B vector
// copy 16 bytes (8 bf16) global -> LDS via the async DMA path (ASYNCcnt)
__device__ __forceinline__ void async_cp16(const u16* gsrc, u16* ldst) {
  __builtin_amdgcn_global_load_async_to_lds_b128((gv4i*)gsrc, (lv4i*)ldst, 0, 0);
}
#define WAIT_ASYNC(n) __builtin_amdgcn_s_wait_asynccnt(n)
#else
__device__ __forceinline__ void async_cp16(const u16* gsrc, u16* ldst) {
  *(uint4*)ldst = *(const uint4*)gsrc;  // all call sites are 16B aligned
}
#define WAIT_ASYNC(n)
#endif

// ---------------- bf16 helpers ----------------
__device__ __forceinline__ u16 f32_to_bf16(float f) {
  u32 u = __builtin_bit_cast(u32, f);
  u32 r = (u + 0x7FFFu + ((u >> 16) & 1u)) >> 16;  // round-to-nearest-even
  return (u16)r;
}
__device__ __forceinline__ float bf16_to_f32(u16 h) {
  return __builtin_bit_cast(float, (u32)h << 16);
}
__device__ __forceinline__ __bf16 bf16_raw(u16 h) {
  return __builtin_bit_cast(__bf16, h);
}

// ---------------- WMMA fragment loaders (CDNA5 ISA 7.12.2) ----------------
// A 16x32 bf16: r=lane&15 (M), hi=lane>>4; elems 0..7 -> K=8hi+i, 8..15 -> K=16+8hi+(i-8)
template<int STRIDE>
__device__ __forceinline__ v16bf load_a_frag(const u16* base, int r, int hi) {
  const u16* p = base + r * STRIDE + 8 * hi;
  v16bf a;
#pragma unroll
  for (int i = 0; i < 8; ++i) a[i] = bf16_raw(p[i]);
#pragma unroll
  for (int i = 0; i < 8; ++i) a[8 + i] = bf16_raw(p[16 + i]);
  return a;
}
// B 32x16 bf16: N = lane&15, hi=lane>>4; elems 0..7 -> K=8hi+i, 8..15 -> K=16+8hi+(i-8)
template<int STRIDE>
__device__ __forceinline__ v16bf load_b_frag(const u16* base, int lane, int nbase) {
  const int hi = lane >> 4;
  const u16* p = base + (8 * hi) * STRIDE + nbase + (lane & 15);
  v16bf b;
#pragma unroll
  for (int i = 0; i < 8; ++i) b[i] = bf16_raw(p[i * STRIDE]);
#pragma unroll
  for (int i = 0; i < 8; ++i) b[8 + i] = bf16_raw(p[(16 + i) * STRIDE]);
  return b;
}
__device__ __forceinline__ v8f wmma_bf16(v16bf a, v16bf b, v8f c) {
  return __builtin_amdgcn_wmma_f32_16x16x32_bf16(false, a, false, b, (short)0, c,
                                                 false, false);
}

// ---------------- sizes ----------------
#define BATCH   1024
#define SEQ     128
#define HID     512
#define GUIDD   256
#define NH      8
#define HD      64
#define MROWS   (BATCH * SEQ)      // 131072
#define KQK     (HID + GUIDD)      // 768

// ---------- kernel 0a: fp32 -> bf16 (weights) ----------
__global__ void convert_f32_bf16(const float* __restrict__ src, u16* __restrict__ dst,
                                 int n) {
  int i = blockIdx.x * blockDim.x + threadIdx.x;
  if (i < n) dst[i] = f32_to_bf16(src[i]);
}

// ---------- kernel 0b: build concatenated bf16 [x | guidance], ld = 768 ----------
__global__ __launch_bounds__(256)
void build_xg_bf16(const float* __restrict__ x, const float* __restrict__ g,
                   u16* __restrict__ xgB) {
  int t = blockIdx.x * blockDim.x + threadIdx.x;  // MROWS*96 threads, 8 elems each
  int row = t / (KQK / 8);
  int cb = (t % (KQK / 8)) * 8;
  const float* src; long idx;
  if (cb < HID) { src = x; idx = (long)row * HID + cb; }
  else          { src = g; idx = (long)row * GUIDD + (cb - HID); }
  float4 f0 = *(const float4*)(src + idx);
  float4 f1 = *(const float4*)(src + idx + 4);
  uint4 w;
  w.x = (u32)f32_to_bf16(f0.x) | ((u32)f32_to_bf16(f0.y) << 16);
  w.y = (u32)f32_to_bf16(f0.z) | ((u32)f32_to_bf16(f0.w) << 16);
  w.z = (u32)f32_to_bf16(f1.x) | ((u32)f32_to_bf16(f1.y) << 16);
  w.w = (u32)f32_to_bf16(f1.z) | ((u32)f32_to_bf16(f1.w) << 16);
  *(uint4*)(xgB + (long)row * KQK + cb) = w;
}

// ---------- kernel 1: fused QKV projection GEMM ----------
// grid = (N/128 = 4, M/64 = 2048, 3 modes), block = 256 (8 waves, 2x4 over 64x128)
// mode 0: Q=elu(xg@Wq+bq)+1 -> Qbuf ; mode 1: K=elu(xg@Wk+bk)+1 -> Kt[b,h,d,l]
// mode 2: V=x@Wv+bv -> Vbuf
__global__ __launch_bounds__(256)
void qkv_gemm_kernel(const u16* __restrict__ xgB, const u16* __restrict__ WqB,
                     const u16* __restrict__ WkB, const u16* __restrict__ WvB,
                     const float* __restrict__ bq, const float* __restrict__ bk,
                     const float* __restrict__ bv, u16* __restrict__ Qbuf,
                     u16* __restrict__ Kt, u16* __restrict__ Vbuf) {
  const int mode = blockIdx.z;
  const u16*   W    = (mode == 0) ? WqB : (mode == 1) ? WkB : WvB;
  const float* bias = (mode == 0) ? bq : (mode == 1) ? bk : bv;
  const int ksteps = (mode == 2) ? (HID / 32) : (KQK / 32);

  __shared__ u16 As[2][64 * 40];    // 64x32, row stride 80B (16B-aligned)
  __shared__ u16 Bs[2][32 * 136];   // 32x128, row stride 272B (16B-aligned)

  const int tid = threadIdx.x;
  const int n0 = blockIdx.x * 128;
  const int m0 = blockIdx.y * 64;
  const int wave = tid >> 5, lane = tid & 31;
  const int wm = wave >> 2, wn = wave & 3;      // 2 x 4 waves
  const int r = lane & 15, hi = lane >> 4;

  // per-thread staging coordinates
  const int arow = tid >> 2, acb = (tid & 3) * 8;    // A: 1x16B per thread
  const int brow = tid >> 3, bcb = (tid & 7) * 16;   // B: 2x16B per thread

  auto issue = [&](int k0, int buf) {
    async_cp16(xgB + (long)(m0 + arow) * KQK + k0 + acb, &As[buf][arow * 40 + acb]);
    const u16* s = W + (long)(k0 + brow) * HID + n0 + bcb;
    u16* d = &Bs[buf][brow * 136 + bcb];
    async_cp16(s, d);
    async_cp16(s + 8, d + 8);
  };

  v8f acc[2][2] = {};
  issue(0, 0);
  for (int i = 0; i < ksteps; ++i) {
    const int cur = i & 1;
    if (i + 1 < ksteps) {
      issue((i + 1) * 32, cur ^ 1);
      WAIT_ASYNC(3);          // step-i tiles complete (in-order ASYNCcnt)
    } else {
      WAIT_ASYNC(0);
    }
    __syncthreads();
    const u16* Ab = As[cur];
    const u16* Bb = Bs[cur];
    v16bf a0 = load_a_frag<40>(Ab + (wm * 32) * 40, r, hi);
    v16bf a1 = load_a_frag<40>(Ab + (wm * 32 + 16) * 40, r, hi);
    v16bf b0 = load_b_frag<136>(Bb, lane, wn * 32);
    v16bf b1 = load_b_frag<136>(Bb, lane, wn * 32 + 16);
    acc[0][0] = wmma_bf16(a0, b0, acc[0][0]);
    acc[0][1] = wmma_bf16(a0, b1, acc[0][1]);
    acc[1][0] = wmma_bf16(a1, b0, acc[1][0]);
    acc[1][1] = wmma_bf16(a1, b1, acc[1][1]);
    __syncthreads();          // protect cur buffer before next issue overwrites it
  }

#pragma unroll
  for (int am = 0; am < 2; ++am)
#pragma unroll
    for (int bn = 0; bn < 2; ++bn)
#pragma unroll
      for (int v = 0; v < 8; ++v) {
        int gm = m0 + wm * 32 + am * 16 + v + 8 * hi;
        int gn = n0 + wn * 32 + bn * 16 + r;
        float val = acc[am][bn][v] + bias[gn];
        if (mode < 2) val = (val > 0.f) ? val + 1.f : __expf(val);  // elu+1
        if (mode == 0) {
          Qbuf[(long)gm * HID + gn] = f32_to_bf16(val);
        } else if (mode == 1) {
          int b = gm >> 7, l = gm & 127;
          int h = gn >> 6, dd = gn & 63;
          Kt[((long)(b * NH + h) * HD + dd) * SEQ + l] = f32_to_bf16(val);
        } else {
          Vbuf[(long)gm * HID + gn] = f32_to_bf16(val);
        }
      }
}

// ---------- kernel 2: KV = K^T @ V (per b,h) + Ksum ----------
__global__ __launch_bounds__(256)
void kv_agg_kernel(const u16* __restrict__ Kt, const u16* __restrict__ Vbuf,
                   u16* __restrict__ KVb, float* __restrict__ Ksum) {
  const int bh = blockIdx.x;
  const int b = bh >> 3, h = bh & 7;
  const int tid = threadIdx.x;

  if (tid < HD) {  // Ksum[d] = sum_l Kt[d,l]
    const u16* p = Kt + (long)bh * (HD * SEQ) + tid * SEQ;
    float s = 0.f;
#pragma unroll 8
    for (int l = 0; l < SEQ; ++l) s += bf16_to_f32(p[l]);
    Ksum[bh * HD + tid] = s;
  }

  __shared__ u16 As[2][64 * 40];
  __shared__ u16 Bs[2][32 * 72];   // row stride 144B (16B-aligned)
  const int wave = tid >> 5, lane = tid & 31;
  const int wm = wave >> 1, wn = wave & 1;
  const int r = lane & 15, hi = lane >> 4;

  const int arow = tid >> 2, acb = (tid & 3) * 8;
  const int brow = tid >> 3, bcb = (tid & 7) * 8;

  auto issue = [&](int k0, int buf) {
    async_cp16(Kt + (long)bh * (HD * SEQ) + arow * SEQ + k0 + acb,
               &As[buf][arow * 40 + acb]);
    async_cp16(Vbuf + (long)(b * SEQ + k0 + brow) * HID + h * HD + bcb,
               &Bs[buf][brow * 72 + bcb]);
  };

  v8f c0 = {}; v8f c1 = {};
  issue(0, 0);
  for (int i = 0; i < 4; ++i) {
    const int cur = i & 1;
    if (i < 3) { issue((i + 1) * 32, cur ^ 1); WAIT_ASYNC(2); }
    else       { WAIT_ASYNC(0); }
    __syncthreads();
    v16bf a  = load_a_frag<40>(&As[cur][wm * 16 * 40], r, hi);
    v16bf b0 = load_b_frag<72>(Bs[cur], lane, wn * 32);
    v16bf b1 = load_b_frag<72>(Bs[cur], lane, wn * 32 + 16);
    c0 = wmma_bf16(a, b0, c0);
    c1 = wmma_bf16(a, b1, c1);
    __syncthreads();
  }
#pragma unroll
  for (int v = 0; v < 8; ++v) {
    int dd = wm * 16 + v + 8 * hi;
    int n0 = wn * 32 + r, n1 = n0 + 16;
    KVb[(long)bh * (HD * HD) + dd * HD + n0] = f32_to_bf16(c0[v]);
    KVb[(long)bh * (HD * HD) + dd * HD + n1] = f32_to_bf16(c1[v]);
  }
}

// ---------- kernel 3: out = Z * (Q @ KV) (per b,h) ----------
__global__ __launch_bounds__(256)
void out_kernel(const u16* __restrict__ Qbuf, const u16* __restrict__ KVb,
                const float* __restrict__ Ksum, float* __restrict__ out) {
  const int bh = blockIdx.x;
  const int b = bh >> 3, h = bh & 7;
  const int tid = threadIdx.x;

  __shared__ u16 As[2][128 * 40];
  __shared__ u16 Bs[64 * 72];
  __shared__ float KsumS[64];
  __shared__ float Zs[128];

  // issue B = whole 64x64 KV tile (2 x 16B per thread)
  {
    int row = tid >> 2, cb = (tid & 3) * 16;
    const u16* s = KVb + (long)bh * (HD * HD) + row * HD + cb;
    u16* d = &Bs[row * 72 + cb];
    async_cp16(s, d);
    async_cp16(s + 8, d + 8);
  }
  // issue A step 0 (128x32)
  const int arow = tid >> 1, acb = (tid & 1) * 16;
  auto issueA = [&](int k0, int buf) {
    const u16* s = Qbuf + (long)(b * SEQ + arow) * HID + h * HD + k0 + acb;
    u16* d = &As[buf][arow * 40 + acb];
    async_cp16(s, d);
    async_cp16(s + 8, d + 8);
  };
  issueA(0, 0);

  if (tid < HD) KsumS[tid] = Ksum[bh * HD + tid];
  __syncthreads();
  if (tid < SEQ) {  // Z[l] = 1/(Q[l]·Ksum + eps)
    const u16* q = Qbuf + (long)(b * SEQ + tid) * HID + h * HD;
    float s = 0.f;
#pragma unroll 8
    for (int d = 0; d < HD; ++d) s += bf16_to_f32(q[d]) * KsumS[d];
    Zs[tid] = 1.f / (s + 1e-6f);
  }

  const int wave = tid >> 5, lane = tid & 31;
  const int r = lane & 15, hi = lane >> 4;
  v8f c0 = {}, c1 = {}, c2 = {}, c3 = {};
  for (int i = 0; i < 2; ++i) {
    const int cur = i & 1;
    if (i == 0) { issueA(32, 1); WAIT_ASYNC(2); }  // B + A0 complete
    else        { WAIT_ASYNC(0); }
    __syncthreads();
    v16bf a  = load_a_frag<40>(&As[cur][wave * 16 * 40], r, hi);
    v16bf b0 = load_b_frag<72>(&Bs[i * 32 * 72], lane, 0);
    v16bf b1 = load_b_frag<72>(&Bs[i * 32 * 72], lane, 16);
    v16bf b2 = load_b_frag<72>(&Bs[i * 32 * 72], lane, 32);
    v16bf b3 = load_b_frag<72>(&Bs[i * 32 * 72], lane, 48);
    c0 = wmma_bf16(a, b0, c0);
    c1 = wmma_bf16(a, b1, c1);
    c2 = wmma_bf16(a, b2, c2);
    c3 = wmma_bf16(a, b3, c3);
    __syncthreads();
  }
#pragma unroll
  for (int v = 0; v < 8; ++v) {
    int l = wave * 16 + v + 8 * hi;
    float z = Zs[l];
    float* o = out + (long)(b * SEQ + l) * HID + h * HD;
    o[r]      = c0[v] * z;
    o[r + 16] = c1[v] * z;
    o[r + 32] = c2[v] * z;
    o[r + 48] = c3[v] * z;
  }
}

// ---------------- host launch ----------------
extern "C" void kernel_launch(void* const* d_in, const int* in_sizes, int n_in,
                              void* d_out, int out_size, void* d_ws, size_t ws_size,
                              hipStream_t stream) {
  const float* x  = (const float*)d_in[0];
  const float* g  = (const float*)d_in[1];
  const float* Wq = (const float*)d_in[2];
  const float* bq = (const float*)d_in[3];
  const float* Wk = (const float*)d_in[4];
  const float* bk = (const float*)d_in[5];
  const float* Wv = (const float*)d_in[6];
  const float* bv = (const float*)d_in[7];
  float* out = (float*)d_out;

  char* ws = (char*)d_ws;
  size_t off = 0;
  auto alloc = [&](size_t bytes) -> void* {
    void* p = ws + off;
    off = (off + bytes + 255) & ~(size_t)255;
    return p;
  };
  const int nWq = KQK * HID, nWv = HID * HID;
  u16* WqB  = (u16*)alloc((size_t)nWq * 2);
  u16* WkB  = (u16*)alloc((size_t)nWq * 2);
  u16* WvB  = (u16*)alloc((size_t)nWv * 2);
  u16* xgB  = (u16*)alloc((size_t)MROWS * KQK * 2);
  u16* Qbuf = (u16*)alloc((size_t)MROWS * HID * 2);
  u16* Kt   = (u16*)alloc((size_t)MROWS * HID * 2);
  u16* Vbuf = (u16*)alloc((size_t)MROWS * HID * 2);
  u16* KVb  = (u16*)alloc((size_t)BATCH * NH * HD * HD * 2);
  float* Ks = (float*)alloc((size_t)BATCH * NH * HD * 4);

  convert_f32_bf16<<<(nWq + 255) / 256, 256, 0, stream>>>(Wq, WqB, nWq);
  convert_f32_bf16<<<(nWq + 255) / 256, 256, 0, stream>>>(Wk, WkB, nWq);
  convert_f32_bf16<<<(nWv + 255) / 256, 256, 0, stream>>>(Wv, WvB, nWv);
  build_xg_bf16<<<(MROWS * (KQK / 8)) / 256, 256, 0, stream>>>(x, g, xgB);

  qkv_gemm_kernel<<<dim3(HID / 128, MROWS / 64, 3), 256, 0, stream>>>(
      xgB, WqB, WkB, WvB, bq, bk, bv, Qbuf, Kt, Vbuf);

  kv_agg_kernel<<<BATCH * NH, 256, 0, stream>>>(Kt, Vbuf, KVb, Ks);

  out_kernel<<<BATCH * NH, 256, 0, stream>>>(Qbuf, KVb, Ks, out);
}
